// DistributionalQNetwork_85452669322027
// MI455X (gfx1250) — matrix operand order repaired and checked by reading, not compiled, not run on hardware
//
#include <hip/hip_runtime.h>

typedef __attribute__((ext_vector_type(16))) _Float16 v16h;
typedef __attribute__((ext_vector_type(8)))  _Float16 v8h;
typedef __attribute__((ext_vector_type(8)))  float    v8f;

#define NOBS   64
#define NACT   16
#define NATOMS 101
#define VMINF  (-10.0f)
#define VMAXF  (10.0f)
#define INV_DZ (5.0f)   /* 1 / ((VMAX-VMIN)/(101-1)) = 1/0.2 */

// Transposed f16 weight layout in scratch / LDS (offsets in halfs):
//  W1t [256][96]  (K 80 -> pad 96)
//  W2t [128][256]
//  W3t [ 64][128]
//  W4t [128][ 64] (N 101 -> pad 128 so NT=8 is 4-tile groupable)
#define W1_OFF  0
#define W2_OFF  24576
#define W3_OFF  57344
#define W4_OFF  65536
#define W_HALFS 73728
#define W_BYTES (W_HALFS * 2)          // 147456

#define ACT_BYTES_PER_WAVE 16384       // two 8KB ping-pong buffers per wave
#define SMEM_BYTES (W_BYTES + 8 * ACT_BYTES_PER_WAVE)   // 278528 <= 320KB WGP LDS

// ---------------------------------------------------------------------------
// Pre-kernel: convert fp32 row-major [K][N] weights into f16 transposed
// [N][Kpad] layout so WMMA B-fragments are two contiguous 8-half runs (k0..k0+7, k0+16..k0+23) per lane.
// ---------------------------------------------------------------------------
__global__ void convert_weights_kernel(const float* __restrict__ W1,
                                       const float* __restrict__ W2,
                                       const float* __restrict__ W3,
                                       const float* __restrict__ W4,
                                       _Float16* __restrict__ ws)
{
    int i = blockIdx.x * blockDim.x + threadIdx.x;
    if (i >= W_HALFS) return;
    if (i < W2_OFF) {                       // W1t [256][96] <- W1 [80][256]
        int n = i / 96, k = i - n * 96;
        ws[i] = (k < 80) ? (_Float16)W1[k * 256 + n] : (_Float16)0.0f;
    } else if (i < W3_OFF) {                // W2t [128][256] <- W2 [256][128]
        int j = i - W2_OFF; int n = j >> 8, k = j & 255;
        ws[i] = (_Float16)W2[k * 128 + n];
    } else if (i < W4_OFF) {                // W3t [64][128] <- W3 [128][64]
        int j = i - W3_OFF; int n = j >> 7, k = j & 127;
        ws[i] = (_Float16)W3[k * 64 + n];
    } else {                                // W4t [128][64] <- W4 [64][101]
        int j = i - W4_OFF; int n = j >> 6, k = j & 63;
        ws[i] = (n < NATOMS) ? (_Float16)W4[k * NATOMS + n] : (_Float16)0.0f;
    }
}

// ---------------------------------------------------------------------------
// WMMA fragment helpers (wave32, 16x16x32 f16 -> f32)
// ---------------------------------------------------------------------------
__device__ __forceinline__ v16h make16(v8h lo, v8h hi) {
    return __builtin_shufflevector(lo, hi, 0,1,2,3,4,5,6,7,8,9,10,11,12,13,14,15);
}

// A (16xK, row-major f16 in LDS).  ISA layout: lanes 0-15 row m=lane hold
// K {kb+0..7, kb+16..23}; lanes 16-31 row m=lane-16 hold K {kb+8..15, kb+24..31}.
__device__ __forceinline__ v16h load_a_frag(const _Float16* base, int ldk, int kb, int lane) {
    int m  = lane & 15;
    int hi = (lane >> 4) * 8;
    const _Float16* row = base + m * ldk + kb + hi;
    v8h x0 = *(const v8h*)(row);
    v8h x1 = *(const v8h*)(row + 16);
    return make16(x0, x1);
}

// B (KxN) from transposed weights Wt[n][k]: lane holds column n=lane&15,
// K {kb+0..7, kb+16..23} (lanes 0-15) / {kb+8..15, kb+24..31} (lanes 16-31).
__device__ __forceinline__ v16h load_b_frag(const _Float16* wt, int ldk, int nt, int kb, int lane) {
    int n  = nt * 16 + (lane & 15);
    int k0 = kb + (lane >> 4) * 8;
    const _Float16* p = wt + n * ldk + k0;
    v8h x0 = *(const v8h*)(p);
    v8h x1 = *(const v8h*)(p + 16);
    return make16(x0, x1);
}

__device__ __forceinline__ v8f wmma16(v16h a, v16h b, v8f c) {
    return __builtin_amdgcn_wmma_f32_16x16x32_f16(false, a, false, b, (short)0, c, false, false);
}

// C layout: VGPR v, lanes 0-15 -> (M=v, N=lane); lanes 16-31 -> (M=v+8, N=lane-16)
template<bool FINAL>
__device__ __forceinline__ void store_tile(v8f c, int nt, const float* __restrict__ bias,
                                           _Float16* hOut, int hLd, float* logitOut, int lane)
{
    int n  = nt * 16 + (lane & 15);
    int m0 = (lane >> 4) * 8;
    if (FINAL) {
        if (n < NATOMS) {
            float bv = bias[n];
#pragma unroll
            for (int v = 0; v < 8; ++v)
                logitOut[(m0 + v) * 112 + n] = c[v] + bv;
        }
    } else {
        float bv = bias[n];
#pragma unroll
        for (int v = 0; v < 8; ++v) {
            float x = c[v] + bv;
            hOut[(m0 + v) * hLd + n] = (_Float16)(x > 0.f ? x : 0.f);
        }
    }
}

// One layer for one wave's 16-row slab.  N-tiles processed 4 at a time so the
// scheduler has 4 independent accumulator chains to interleave (hides the
// ~5-slot f16 WMMA->WMMA accumulation hazard instead of eating v_nops).
template<int KT, int NT, bool FINAL>
__device__ __forceinline__ void wave_layer(const _Float16* aBuf, int aLd,
                                           const _Float16* wt,  int wLd,
                                           const float* __restrict__ bias,
                                           _Float16* hOut, int hLd,
                                           float* logitOut, int lane)
{
    static_assert(NT % 4 == 0, "NT must be a multiple of 4");
    v16h a[KT];
#pragma unroll
    for (int ks = 0; ks < KT; ++ks) a[ks] = load_a_frag(aBuf, aLd, ks * 32, lane);

#pragma unroll
    for (int ng = 0; ng < NT; ng += 4) {
        v8f c0 = {0.f,0.f,0.f,0.f,0.f,0.f,0.f,0.f};
        v8f c1 = c0, c2 = c0, c3 = c0;
#pragma unroll
        for (int ks = 0; ks < KT; ++ks) {
            v16h b0 = load_b_frag(wt, wLd, ng + 0, ks * 32, lane);
            v16h b1 = load_b_frag(wt, wLd, ng + 1, ks * 32, lane);
            v16h b2 = load_b_frag(wt, wLd, ng + 2, ks * 32, lane);
            v16h b3 = load_b_frag(wt, wLd, ng + 3, ks * 32, lane);
            c0 = wmma16(a[ks], b0, c0);
            c1 = wmma16(a[ks], b1, c1);
            c2 = wmma16(a[ks], b2, c2);
            c3 = wmma16(a[ks], b3, c3);
        }
        store_tile<FINAL>(c0, ng + 0, bias, hOut, hLd, logitOut, lane);
        store_tile<FINAL>(c1, ng + 1, bias, hOut, hLd, logitOut, lane);
        store_tile<FINAL>(c2, ng + 2, bias, hOut, hLd, logitOut, lane);
        store_tile<FINAL>(c3, ng + 3, bias, hOut, hLd, logitOut, lane);
    }
}

// ---------------------------------------------------------------------------
// Fused MLP + softmax + C51 projection.  256 threads = 8 waves, 16 rows/wave.
// ---------------------------------------------------------------------------
__global__ __launch_bounds__(256)
void c51_fused_kernel(const float* __restrict__ obs, const float* __restrict__ act,
                      const float* __restrict__ rewards, const float* __restrict__ bootstrap,
                      const float* __restrict__ discount, const float* __restrict__ qsup,
                      const float* __restrict__ b1, const float* __restrict__ b2,
                      const float* __restrict__ b3, const float* __restrict__ b4,
                      const _Float16* __restrict__ gw, float* __restrict__ out, int B)
{
    extern __shared__ char smem[];
    _Float16* wl = (_Float16*)smem;

    const int lane = threadIdx.x & 31;
    const int wave = threadIdx.x >> 5;
    const int slab = (blockIdx.x * 8 + wave) * 16;

    // Stage all (transposed f16) weights into LDS via async LDS DMA
    // (GLOBAL_LOAD_ASYNC_TO_LDS_B128, ASYNCcnt) - no VGPR round trip, and it
    // overlaps with the x-staging below.
    {
        const unsigned long long gbase = (unsigned long long)(uintptr_t)gw;
        const unsigned ldsBase = (unsigned)(uintptr_t)smem;   // flat addr[31:0] == LDS offset
        for (int i = threadIdx.x; i < (W_BYTES / 16); i += 256) {
            unsigned voff = (unsigned)i * 16u;
            unsigned doff = ldsBase + voff;
            asm volatile("global_load_async_to_lds_b128 %0, %1, %2"
                         :: "v"(doff), "v"(voff), "s"(gbase)
                         : "memory");
        }
    }

    char* aw = smem + W_BYTES + wave * ACT_BYTES_PER_WAVE;
    _Float16* buf0 = (_Float16*)aw;            // 8KB ping
    _Float16* buf1 = (_Float16*)(aw + 8192);   // 8KB pong

    // Stage x = concat(obs, actions) as f16, K padded 80 -> 96 with zeros.
    if (slab < B) {
        for (int i = lane; i < 16 * 96; i += 32) {
            int r = i / 96, c = i - r * 96;
            int g = slab + r;
            float v = 0.0f;
            if (c < NOBS)             v = obs[g * NOBS + c];
            else if (c < NOBS + NACT) v = act[g * NACT + (c - NOBS)];
            buf0[i] = (_Float16)v;
        }
    }

    asm volatile("s_wait_asynccnt 0x0" ::: "memory");
    __syncthreads();
    if (slab >= B) return;

    const _Float16* W1t = wl + W1_OFF;
    const _Float16* W2t = wl + W2_OFF;
    const _Float16* W3t = wl + W3_OFF;
    const _Float16* W4t = wl + W4_OFF;

    // L1: [16,96]x[96,256]   -> h1 (buf1)     48 WMMA
    wave_layer<3, 16, false>(buf0, 96,  W1t, 96,  b1, buf1, 256, nullptr, lane);
    // L2: [16,256]x[256,128] -> h2 (buf0)     64 WMMA
    wave_layer<8,  8, false>(buf1, 256, W2t, 256, b2, buf0, 128, nullptr, lane);
    // L3: [16,128]x[128,64]  -> h3 (buf1)     16 WMMA
    wave_layer<4,  4, false>(buf0, 128, W3t, 128, b3, buf1, 64,  nullptr, lane);
    // L4: [16,64]x[64,128]   -> logits f32 (buf0, stride 112)   16 WMMA
    float* lg = (float*)buf0;
    wave_layer<2,  8, true >(buf1, 64,  W4t, 64,  b4, nullptr, 0, lg, lane);

    // Softmax + C51 categorical projection; lane r < 16 owns row r.
    float* pr = (float*)buf1;   // proj [16][101] f32
    if (lane < 16) {
        const int r = lane, g = slab + r;
        float* L = lg + r * 112;

        float mx = -3.0e38f;
        for (int j = 0; j < NATOMS; ++j) mx = fmaxf(mx, L[j]);
        float s = 0.f;
        for (int j = 0; j < NATOMS; ++j) { float e = __expf(L[j] - mx); L[j] = e; s += e; }
        float inv = 1.0f / s;

        float rw = rewards[g];
        float bd = bootstrap[g] * discount[g];
        float* P = pr + r * NATOMS;
        for (int j = 0; j < NATOMS; ++j) P[j] = 0.f;
        for (int j = 0; j < NATOMS; ++j) {
            float p  = L[j] * inv;
            float tz = rw + bd * qsup[j];
            tz = fminf(fmaxf(tz, VMINF), VMAXF);
            float bc = (tz - VMINF) * INV_DZ;
            int lo = (int)floorf(bc);
            int up = (int)ceilf(bc);
            if (lo == up) { if (lo > 0) --lo; else ++up; }   // reference edge handling
            P[lo] += p * ((float)up - bc);
            P[up] += p * (bc - (float)lo);
        }
    }

    // Cooperative coalesced store of the projected distribution.
    for (int i = lane; i < 16 * NATOMS; i += 32) {
        int r = i / NATOMS, c = i - r * NATOMS;
        out[(slab + r) * NATOMS + c] = pr[r * NATOMS + c];
    }
}

// ---------------------------------------------------------------------------
extern "C" void kernel_launch(void* const* d_in, const int* in_sizes, int n_in,
                              void* d_out, int out_size, void* d_ws, size_t ws_size,
                              hipStream_t stream)
{
    (void)n_in; (void)out_size; (void)ws_size;
    const float* obs       = (const float*)d_in[0];
    const float* actions   = (const float*)d_in[1];
    const float* rewards   = (const float*)d_in[2];
    const float* bootstrap = (const float*)d_in[3];
    const float* discount  = (const float*)d_in[4];
    const float* qsup      = (const float*)d_in[5];
    const float* W1 = (const float*)d_in[6];
    const float* b1 = (const float*)d_in[7];
    const float* W2 = (const float*)d_in[8];
    const float* b2 = (const float*)d_in[9];
    const float* W3 = (const float*)d_in[10];
    const float* b3 = (const float*)d_in[11];
    const float* W4 = (const float*)d_in[12];
    const float* b4 = (const float*)d_in[13];
    float*     out = (float*)d_out;
    _Float16*  ws  = (_Float16*)d_ws;

    const int B = in_sizes[0] / NOBS;

    convert_weights_kernel<<<(W_HALFS + 255) / 256, 256, 0, stream>>>(W1, W2, W3, W4, ws);

    (void)hipFuncSetAttribute((const void*)c51_fused_kernel,
                              hipFuncAttributeMaxDynamicSharedMemorySize,
                              (int)SMEM_BYTES);
    const int nblk = (B + 127) / 128;
    c51_fused_kernel<<<nblk, 256, SMEM_BYTES, stream>>>(
        obs, actions, rewards, bootstrap, discount, qsup,
        b1, b2, b3, b4, ws, out, B);
}
